// VQVAEQuantizer_41162966565038
// MI455X (gfx1250) — matrix operand order repaired and checked by reading, not compiled, not run on hardware
//
#include <hip/hip_runtime.h>
#include <hip/hip_bf16.h>

typedef __attribute__((ext_vector_type(16))) __bf16 v16bf;
typedef __attribute__((ext_vector_type(8)))  float  v8f;

#define C_DIM   256
#define K_CODES 1024
#define N_ROWS  32768           // 32 * 32 * 32
#define TOTAL_ELEMS 8388608     // 32 * 256 * 32 * 32

static __device__ __forceinline__ unsigned short f2bf(float f) {
  unsigned int u = __float_as_uint(f);
  u += 0x7FFFu + ((u >> 16) & 1u);   // round-to-nearest-even
  return (unsigned short)(u >> 16);
}

// ---------------- Kernel 1: per-code squared norms ----------------
__global__ __launch_bounds__(256) void vq_e2_kernel(const float* __restrict__ emb,
                                                    float* __restrict__ e2g) {
  __shared__ float red[256];
  const int n = blockIdx.x, t = threadIdx.x;
  float v = emb[n * C_DIM + t];
  red[t] = v * v;
  __syncthreads();
  for (int s = 128; s > 0; s >>= 1) {
    if (t < s) red[t] += red[t + s];
    __syncthreads();
  }
  if (t == 0) e2g[n] = red[0];
}

// Stage one B tile (C=256 x 16 codes) as bf16 into LDS in WMMA register
// order: element e[n, c] -> buf[(cg*16 + n)*16 + (c & 15)], which equals the
// reader's [(kc*32 + hi16*16 + n)*16 + i] with c = kc*32 + hi16*16 + i.
static __device__ __forceinline__ void stageB(const float* __restrict__ emb,
                                              unsigned short* __restrict__ buf,
                                              int t, int n_local, int cg) {
  const float* erow = emb + ((size_t)(t * 16 + n_local) * C_DIM) + cg * 16;
  unsigned short* dst = &buf[(cg * 16 + n_local) * 16];
  #pragma unroll
  for (int cc = 0; cc < 16; ++cc) dst[cc] = f2bf(erow[cc]);
}

// ---------------- Kernel 2: WMMA distance + argmin ----------------
// Block = 256 threads = 8 waves; each wave owns TWO 16-row tiles (32 rows) of
// flat_x so every B fragment loaded from LDS feeds 2 WMMAs (halves LDS BW per
// MAC). B tiles are double-buffered in LDS: stage t+1 while computing t.
__global__ __launch_bounds__(256) void vq_argmin_kernel(const float* __restrict__ x,
                                                        const float* __restrict__ emb,
                                                        const float* __restrict__ e2g,
                                                        int* __restrict__ idx_out) {
  __shared__ __align__(16) unsigned short ldsB[2][4096];   // 2 x 8 KB ping-pong

  const int tid  = threadIdx.x;
  const int lane = tid & 31;
  const int wave = tid >> 5;
  const int hi   = lane >> 4;                  // half-wave selector
  const int rowBase = blockIdx.x * 256 + wave * 32;

  // Two register-resident A tiles: rows [rowBase, +16) and [rowBase+16, +32).
  // A layout used here (B matches): lane(0-15)=row; element i -> K = i + hi*16.
  v16bf a0[8], a1[8];
  #pragma unroll
  for (int half = 0; half < 2; ++half) {
    const int m  = rowBase + half * 16 + (lane & 15);
    const int b  = m >> 10;                    // row / (H*W)
    const int hw = m & 1023;
    const float* xrow = x + (((size_t)b * C_DIM) << 10) + hw;  // stride c = 1024 f
    #pragma unroll
    for (int kc = 0; kc < 8; ++kc) {
      union { v16bf v; unsigned short s[16]; } ua;
      const int cbase = kc * 32 + hi * 16;
      #pragma unroll
      for (int i = 0; i < 16; ++i) {
        const int c = cbase + i;
        ua.s[i] = f2bf(xrow[(size_t)c << 10]);
      }
      if (half == 0) a0[kc] = ua.v; else a1[kc] = ua.v;
    }
  }

  float minv0[8], minv1[8];
  int   mini0[8], mini1[8];
  #pragma unroll
  for (int r = 0; r < 8; ++r) {
    minv0[r] = 3.4e38f; mini0[r] = 0;
    minv1[r] = 3.4e38f; mini1[r] = 0;
  }

  const int n_local = tid >> 4;  // staging: which code row (0..15)
  const int cg      = tid & 15;  // staging: which 16-wide c group

  stageB(emb, ldsB[0], 0, n_local, cg);
  __syncthreads();

  for (int t = 0; t < 64; ++t) {
    const unsigned short* cur = ldsB[t & 1];
    if (t + 1 < 64) stageB(emb, ldsB[(t + 1) & 1], t + 1, n_local, cg);

    // 8 K-steps; each B fragment feeds two independent WMMA chains.
    v8f acc0 = {}, acc1 = {};
    #pragma unroll
    for (int kc = 0; kc < 8; ++kc) {
      v16bf bvec = *(const v16bf*)&cur[(kc * 32 + lane) * 16];
      acc0 = __builtin_amdgcn_wmma_f32_16x16x32_bf16(false, a0[kc], false, bvec,
                                                     (short)0, acc0, false, false);
      acc1 = __builtin_amdgcn_wmma_f32_16x16x32_bf16(false, a1[kc], false, bvec,
                                                     (short)0, acc1, false, false);
    }

    // score = ||e_n||^2 - 2 * <x_m, e_n>   (||x||^2 constant per row)
    const int   n   = t * 16 + (lane & 15);    // D layout: lane -> column
    const float e2n = e2g[n];
    #pragma unroll
    for (int r = 0; r < 8; ++r) {              // D layout: vgpr r -> row r + hi*8
      const float s0 = e2n - 2.0f * acc0[r];
      const float s1 = e2n - 2.0f * acc1[r];
      if (s0 < minv0[r]) { minv0[r] = s0; mini0[r] = n; }
      if (s1 < minv1[r]) { minv1[r] = s1; mini1[r] = n; }
    }
    __syncthreads();
  }

  // Reduce each row's (min, argmin) across the 16 lanes holding its columns.
  #pragma unroll
  for (int half = 0; half < 2; ++half) {
    #pragma unroll
    for (int r = 0; r < 8; ++r) {
      float v = half ? minv1[r] : minv0[r];
      int  ix = half ? mini1[r] : mini0[r];
      #pragma unroll
      for (int msk = 1; msk < 16; msk <<= 1) { // stays within each 16-lane half
        const float ov = __shfl_xor(v, msk, 32);
        const int   oi = __shfl_xor(ix, msk, 32);
        if (ov < v || (ov == v && oi < ix)) { v = ov; ix = oi; }
      }
      const int base = rowBase + half * 16;
      if (lane == r)      idx_out[base + r]     = ix;   // rows 0..7  (hi=0)
      if (lane == 16 + r) idx_out[base + 8 + r] = ix;   // rows 8..15 (hi=1)
    }
  }
}

// ---------------- Kernel 3: gather codes + squared-error partials ----------------
__global__ __launch_bounds__(256) void vq_quant_kernel(const float* __restrict__ x,
                                                       const float* __restrict__ emb,
                                                       const int* __restrict__ idx,
                                                       float* __restrict__ out,
                                                       float* __restrict__ partials) {
  const int t = threadIdx.x;
  const size_t lin = (size_t)blockIdx.x * 256 + t;   // linear index in [B,C,H,W]
  const int b  = (int)(lin >> 18);                   // / (C*H*W) = / 262144
  const int c  = (int)((lin >> 10) & 255);
  const int hw = (int)(lin & 1023);
  const int row = (b << 10) + hw;
  const int id  = idx[row];
  const float q  = emb[(size_t)id * C_DIM + c];
  const float xv = x[lin];
  out[lin] = q;                                      // straight-through == quantized
  float d = q - xv; d *= d;
  __shared__ float red[256];
  red[t] = d;
  __syncthreads();
  for (int s = 128; s > 0; s >>= 1) {
    if (t < s) red[t] += red[t + s];
    __syncthreads();
  }
  if (t == 0) partials[blockIdx.x] = red[0];
}

// ---------------- Kernel 4: deterministic final loss reduction ----------------
__global__ __launch_bounds__(256) void vq_loss_kernel(const float* __restrict__ partials,
                                                      float* __restrict__ lossOut) {
  const int t = threadIdx.x;
  float s = 0.0f;
  for (int i = t; i < N_ROWS; i += 256) s += partials[i];  // fixed order
  __shared__ float red[256];
  red[t] = s;
  __syncthreads();
  for (int st = 128; st > 0; st >>= 1) {
    if (t < st) red[t] += red[t + st];
    __syncthreads();
  }
  // loss = q_latent + 0.25*e_latent, both numerically mean((q - x)^2)
  if (t == 0) *lossOut = red[0] * (1.25f / (float)TOTAL_ELEMS);
}

extern "C" void kernel_launch(void* const* d_in, const int* in_sizes, int n_in,
                              void* d_out, int out_size, void* d_ws, size_t ws_size,
                              hipStream_t stream) {
  const float* x   = (const float*)d_in[0];   // [32, 256, 32, 32]
  const float* emb = (const float*)d_in[1];   // [1024, 256]
  float* out = (float*)d_out;                 // [32, 256, 32, 32] ++ [loss]
  float* lossOut = out + (out_size - 1);

  char* ws = (char*)d_ws;
  float* e2g      = (float*)(ws);               // 1024 f  (4 KB)
  int*   idx      = (int*)  (ws + 4096);        // 32768 i (128 KB)
  float* partials = (float*)(ws + 135168);      // 32768 f (128 KB)

  vq_e2_kernel    <<<K_CODES,           256, 0, stream>>>(emb, e2g);
  vq_argmin_kernel<<<N_ROWS / 256,      256, 0, stream>>>(x, emb, e2g, idx);
  vq_quant_kernel <<<TOTAL_ELEMS / 256, 256, 0, stream>>>(x, emb, idx, out, partials);
  vq_loss_kernel  <<<1,                 256, 0, stream>>>(partials, lossOut);
}
